// DynamicFormer_90572270338248
// MI455X (gfx1250) — matrix operand order repaired and checked by reading, not compiled, not run on hardware
//
#include <hip/hip_runtime.h>
#include <math.h>

// ---------------- model constants ----------------
#define BBATCH 4
#define LL     192
#define DIMD   384
#define BSR    768           // BBATCH*LL
#define H2C    24            // 2*HEADS
#define NHH    12            // HEADS
#define HDD    16            // diff-attn head dim
#define VDD    32            // v head dim
#define FREQC  256
#define IMHW   2304          // 48*48
#define NTOKD  294912L       // BSR*DIMD
#define NPAIR  3538944L      // BBATCH*LL*LL*H2C
#define NATT   1769472L      // BBATCH*NHH*LL*LL

typedef __attribute__((ext_vector_type(16))) _Float16 v16h;
typedef __attribute__((ext_vector_type(8)))  float    v8f;
typedef __attribute__((ext_vector_type(4)))  float    v4f;

// =====================================================================
// Generic batched/strided WMMA GEMM:  D = alpha * (A @ opB(B)) (+ bias[n])
// REQUIRES: M % 32 == 0, N % 32 == 0, K % 16 == 0 (true for every call here;
// K < 32 handled by whole-segment zero padding, only in the GUARD variant).
//   A:(M,K)  element (m,k) at A + zo*aO + zi*aI + m*lda + k
//   B TRANSB=1: element (n,k) at B + zo*bO + zi*bI + n*ldb + k   (x @ W^T)
//   B TRANSB=0: element (k,n) at B + zo*bO + zi*bI + k*ldb + n
//   D element (m,n) at D + zo*dO + zi*dI + m*dRow + n*dCol
// One wave computes a 32x32 macro-tile: 2 A frags x 2 B frags -> 4 WMMAs/step.
// TRANSB / GUARD are compile-time so the unused address path and the K-tail
// branches are fully eliminated in the hot instantiations.
// =====================================================================
struct GemmP {
  const float* A; const float* Bm; float* D; const float* bias;
  long aO, aI, bO, bI, dO, dI, dRow;
  int inner, lda, ldb, dCol, M, N, K, hasBias;
  float alpha;
};

// A fragment (16x32, f16): element e -> K = half*16 + e, row = l16
template<bool GUARD>
__device__ __forceinline__ v16h load_a_frag(const float* __restrict__ arow,
                                            int k0, int half, int K) {
  v16h a;
  #pragma unroll
  for (int seg = 0; seg < 2; ++seg) {
    const int kb = k0 + half * 16 + seg * 8;
    if (!GUARD || (kb + 8 <= K)) {
      v4f a0 = *(const v4f*)(arow + kb);
      v4f a1 = *(const v4f*)(arow + kb + 4);
      #pragma unroll
      for (int j = 0; j < 4; ++j) {
        a[seg*8 + j]     = (_Float16)a0[j];
        a[seg*8 + 4 + j] = (_Float16)a1[j];
      }
    } else {
      #pragma unroll
      for (int j = 0; j < 8; ++j) a[seg*8 + j] = (_Float16)0.f;
    }
  }
  return a;
}

// B fragment (32x16, f16), B stored (N,K): element e -> K = half*16 + e, col = l16
template<bool GUARD>
__device__ __forceinline__ v16h load_b_frag_t(const float* __restrict__ brow,
                                              int k0, int half, int K) {
  v16h b;
  const int kb = k0 + half * 16;
  if (!GUARD || (kb + 16 <= K)) {
    #pragma unroll
    for (int q = 0; q < 4; ++q) {
      v4f t = *(const v4f*)(brow + kb + q*4);
      #pragma unroll
      for (int j = 0; j < 4; ++j) b[q*4 + j] = (_Float16)t[j];
    }
  } else {
    #pragma unroll
    for (int j = 0; j < 16; ++j) b[j] = (_Float16)0.f;
  }
  return b;
}

// B fragment, B stored (K,N) row-major: strided loads down column l16
template<bool GUARD>
__device__ __forceinline__ v16h load_b_frag_n(const float* __restrict__ bcol,
                                              int k0, int half, int K, int ldb) {
  v16h b;
  const int kb = k0 + half * 16;
  if (!GUARD || (kb + 16 <= K)) {
    #pragma unroll
    for (int j = 0; j < 16; ++j) b[j] = (_Float16)bcol[(long)(kb + j) * ldb];
  } else {
    #pragma unroll
    for (int j = 0; j < 16; ++j) b[j] = (_Float16)0.f;
  }
  return b;
}

template<bool TRANSB, bool GUARD>
__global__ __launch_bounds__(32) void k_gemm(GemmP p) {
  const int lane = threadIdx.x & 31;
  const int half = lane >> 4;
  const int l16  = lane & 15;
  const long tm  = (long)blockIdx.x * 32;
  const long tnn = (long)blockIdx.y * 32;
  const int z    = blockIdx.z;
  const int zo = z / p.inner, zi = z % p.inner;
  const float* Ab = p.A  + zo * p.aO + zi * p.aI;
  const float* Bb = p.Bm + zo * p.bO + zi * p.bI;
  float*       Db = p.D  + zo * p.dO + zi * p.dI;

  const float* arow0 = Ab + (tm + l16) * p.lda;
  const float* arow1 = arow0 + 16L * p.lda;
  const float* brow0 = Bb + (tnn + l16) * p.ldb;  // TRANSB path
  const float* brow1 = brow0 + 16L * p.ldb;
  const float* bcol0 = Bb + tnn + l16;            // !TRANSB path
  const float* bcol1 = bcol0 + 16;

  v8f acc00 = {0.f,0.f,0.f,0.f,0.f,0.f,0.f,0.f};
  v8f acc01 = acc00, acc10 = acc00, acc11 = acc00;

  for (int k0 = 0; k0 < p.K; k0 += 32) {
    const v16h a0 = load_a_frag<GUARD>(arow0, k0, half, p.K);
    const v16h a1 = load_a_frag<GUARD>(arow1, k0, half, p.K);
    v16h b0, b1;
    if (TRANSB) {
      b0 = load_b_frag_t<GUARD>(brow0, k0, half, p.K);
      b1 = load_b_frag_t<GUARD>(brow1, k0, half, p.K);
    } else {
      b0 = load_b_frag_n<GUARD>(bcol0, k0, half, p.K, p.ldb);
      b1 = load_b_frag_n<GUARD>(bcol1, k0, half, p.K, p.ldb);
    }
    acc00 = __builtin_amdgcn_wmma_f32_16x16x32_f16(false, a0, false, b0, (short)0, acc00, false, false);
    acc01 = __builtin_amdgcn_wmma_f32_16x16x32_f16(false, a0, false, b1, (short)0, acc01, false, false);
    acc10 = __builtin_amdgcn_wmma_f32_16x16x32_f16(false, a1, false, b0, (short)0, acc10, false, false);
    acc11 = __builtin_amdgcn_wmma_f32_16x16x32_f16(false, a1, false, b1, (short)0, acc11, false, false);
  }

  // store: D element r of a 16x16 tile -> row base + r + half*8, col base + l16
  const float bs0 = p.hasBias ? p.bias[tnn + l16]      : 0.f;
  const float bs1 = p.hasBias ? p.bias[tnn + 16 + l16] : 0.f;
  #pragma unroll
  for (int r = 0; r < 8; ++r) {
    const long m0 = tm + r + half * 8;
    const long m1 = m0 + 16;
    const long n0 = tnn + l16;
    const long n1 = n0 + 16;
    Db[m0 * p.dRow + n0 * p.dCol] = p.alpha * acc00[r] + bs0;
    Db[m0 * p.dRow + n1 * p.dCol] = p.alpha * acc01[r] + bs1;
    Db[m1 * p.dRow + n0 * p.dCol] = p.alpha * acc10[r] + bs0;
    Db[m1 * p.dRow + n1 * p.dCol] = p.alpha * acc11[r] + bs1;
  }
}

// =====================================================================
// Fused conv stem: per-image workgroup, conv1 kept as f16 in 228KB LDS
// (CDNA5 WGP has 320KB), then conv2+relu+3x3 avg-pool -> pooled (768,4096)
// =====================================================================
#define CONV_SMEM_BYTES 228096   // (6*2304 + 1728 + 4608)*4 + 32*2304*2

__global__ __launch_bounds__(256) void k_conv(const float* __restrict__ x,
    const float* __restrict__ w1, const float* __restrict__ b1,
    const float* __restrict__ w2, const float* __restrict__ b2,
    float* __restrict__ pooled) {
  extern __shared__ unsigned char smem[];
  float*    inp = (float*)smem;            // 6*2304 f32
  float*    w1s = inp + 6*IMHW;            // 1728
  float*    w2s = w1s + 1728;              // 4608
  _Float16* c1  = (_Float16*)(w2s + 4608); // 32*2304 f16

  const int img = blockIdx.x;
  const float* xi = x + (long)img * 6 * IMHW;
  for (int i = threadIdx.x; i < 6*IMHW; i += 256) inp[i] = xi[i];
  for (int i = threadIdx.x; i < 1728;   i += 256) w1s[i] = w1[i];
  for (int i = threadIdx.x; i < 4608;   i += 256) w2s[i] = w2[i];
  __syncthreads();

  // conv1 + relu -> LDS (f16)
  for (int it = threadIdx.x; it < 32*IMHW; it += 256) {
    const int co = it / IMHW, p = it % IMHW;
    const int yy = p / 48, xx = p % 48;
    float s = b1[co];
    for (int ci = 0; ci < 6; ++ci) {
      const float* wp = w1s + (co*6 + ci)*9;
      const float* ip = inp + ci*IMHW;
      #pragma unroll
      for (int dy = 0; dy < 3; ++dy) {
        const int iy = yy + dy - 1;
        if ((unsigned)iy >= 48u) continue;
        #pragma unroll
        for (int dx = 0; dx < 3; ++dx) {
          const int ix = xx + dx - 1;
          if ((unsigned)ix >= 48u) continue;
          s += ip[iy*48 + ix] * wp[dy*3 + dx];
        }
      }
    }
    c1[it] = (_Float16)fmaxf(s, 0.f);
  }
  __syncthreads();

  // conv2 + relu + 3x3 mean-pool: one pool cell per thread
  const int py = threadIdx.x >> 4, px = threadIdx.x & 15;
  for (int co = 0; co < 16; ++co) {
    float acc = 0.f;
    for (int sy = 0; sy < 3; ++sy) {
      for (int sx = 0; sx < 3; ++sx) {
        const int yy = py*3 + sy, xx = px*3 + sx;
        float s = b2[co];
        for (int ci = 0; ci < 32; ++ci) {
          const float* wp = w2s + (co*32 + ci)*9;
          const _Float16* cp = c1 + ci*IMHW;
          #pragma unroll
          for (int dy = 0; dy < 3; ++dy) {
            const int iy = yy + dy - 1;
            if ((unsigned)iy >= 48u) continue;
            #pragma unroll
            for (int dx = 0; dx < 3; ++dx) {
              const int ix = xx + dx - 1;
              if ((unsigned)ix >= 48u) continue;
              s += (float)cp[iy*48 + ix] * wp[dy*3 + dx];
            }
          }
        }
        acc += fmaxf(s, 0.f);
      }
    }
    pooled[(long)img*4096 + co*256 + py*16 + px] = acc * (1.f/9.f);
  }
}

// =====================================================================
// Row-wise RMS / LN : one wave32 per row, shuffle reductions
// =====================================================================
__global__ __launch_bounds__(256) void k_rms(const float* x, float* y,
                                             int rows, int D, float scale) {
  const int wid = blockIdx.x * 8 + (threadIdx.x >> 5);
  const int lane = threadIdx.x & 31;
  if (wid >= rows) return;
  const float* xr = x + (long)wid * D;
  float s = 0.f;
  for (int i = lane; i < D; i += 32) { const float v = xr[i]; s += v*v; }
  for (int o = 16; o > 0; o >>= 1) s += __shfl_down(s, o, 32);
  s = __shfl(s, 0, 32);
  const float r = rsqrtf(s / (float)D + 1e-5f) * scale;
  float* yr = y + (long)wid * D;
  for (int i = lane; i < D; i += 32) yr[i] = xr[i] * r;
}

__global__ __launch_bounds__(256) void k_ln(const float* x, float* y,
                                            const float* g, const float* b,
                                            int rows, int D) {
  const int wid = blockIdx.x * 8 + (threadIdx.x >> 5);
  const int lane = threadIdx.x & 31;
  if (wid >= rows) return;
  const float* xr = x + (long)wid * D;
  float s1 = 0.f, s2 = 0.f;
  for (int i = lane; i < D; i += 32) { const float v = xr[i]; s1 += v; s2 += v*v; }
  for (int o = 16; o > 0; o >>= 1) { s1 += __shfl_down(s1, o, 32); s2 += __shfl_down(s2, o, 32); }
  s1 = __shfl(s1, 0, 32); s2 = __shfl(s2, 0, 32);
  const float mean = s1 / (float)D;
  const float var  = s2 / (float)D - mean * mean;
  const float rs   = rsqrtf(var + 1e-5f);
  float* yr = y + (long)wid * D;
  for (int i = lane; i < D; i += 32) yr[i] = (xr[i] - mean) * rs * g[i] + b[i];
}

// softmax over j of  s[b,c,i,j] + tn[b,i,j,c]   (in-place on s), wave per row
__global__ __launch_bounds__(256) void k_softmax(float* sm, const float* tn) {
  const int wid = blockIdx.x * 8 + (threadIdx.x >> 5);
  const int lane = threadIdx.x & 31;
  const int rows = BBATCH * H2C * LL;
  if (wid >= rows) return;
  const int b = wid / (H2C * LL);
  const int r = wid % (H2C * LL);
  const int c = r / LL;
  const int i = r % LL;
  float* sr = sm + (long)wid * LL;
  const float* tb = tn + ((long)(b * LL + i) * LL) * H2C + c;  // stride H2C over j
  float mx = -1e30f;
  for (int j = lane; j < LL; j += 32) mx = fmaxf(mx, sr[j] + tb[(long)j * H2C]);
  for (int o = 16; o > 0; o >>= 1) mx = fmaxf(mx, __shfl_down(mx, o, 32));
  mx = __shfl(mx, 0, 32);
  float sum = 0.f;
  for (int j = lane; j < LL; j += 32) {
    const float e = expf(sr[j] + tb[(long)j * H2C] - mx);
    sr[j] = e; sum += e;
  }
  for (int o = 16; o > 0; o >>= 1) sum += __shfl_down(sum, o, 32);
  sum = __shfl(sum, 0, 32);
  const float inv = 1.f / sum;
  for (int j = lane; j < LL; j += 32) sr[j] *= inv;
}

// ----------------------------- elementwise -----------------------------
__global__ void k_timestep(const float* t, float* tf) {
  const long idx = (long)blockIdx.x * 256 + threadIdx.x;
  if (idx >= (long)BSR * FREQC) return;
  const int f = (int)(idx % FREQC);
  const int tok = (int)(idx / FREQC);
  const float fr = expf(-9.210340371976184f * (float)(f & 127) / 128.f);
  const float a = t[tok] * fr;
  tf[idx] = (f < 128) ? cosf(a) : sinf(a);
}

__global__ void k_silu(float* x, long n) {
  const long idx = (long)blockIdx.x * 256 + threadIdx.x;
  if (idx >= n) return;
  const float v = x[idx];
  x[idx] = v / (1.f + expf(-v));
}

__global__ void k_add(float* dst, const float* a, const float* b, long n) {
  const long idx = (long)blockIdx.x * 256 + threadIdx.x;
  if (idx >= n) return;
  dst[idx] = a[idx] + b[idx];
}

__global__ void k_swiglu(const float* y, const float* s1, const float* g,
                         float* h, long n) {
  const long idx = (long)blockIdx.x * 256 + threadIdx.x;
  if (idx >= n) return;
  const float gv = g[idx];
  h[idx] = y[idx] + s1[idx] * (gv / (1.f + expf(-gv)));
}

__global__ void k_pair_init(const float* t, const float* pl, const float* pr,
                            float* tp, long n) {
  const long idx = (long)blockIdx.x * 256 + threadIdx.x;
  if (idx >= n) return;
  const int c = (int)(idx % H2C);
  long r = idx / H2C;
  const int j = (int)(r % LL); r /= LL;
  const int i = (int)(r % LL);
  const int b = (int)(r / LL);
  tp[idx] = t[b*LL + j] * pl[c] + t[b*LL + i] * pr[c];
}

__global__ void k_diff(const float* sm, const float* lam, float* attn, long n) {
  const long idx = (long)blockIdx.x * 256 + threadIdx.x;
  if (idx >= n) return;
  const int j = (int)(idx % LL);
  long r = idx / LL;
  const int i  = (int)(r % LL); r /= LL;
  const int hh = (int)(r % NHH);
  const int b  = (int)(r / NHH);
  const long base = (((long)(b*H2C + 2*hh)) * LL + i) * LL + j;
  attn[idx] = sm[base] - lam[0] * sm[base + (long)LL * LL];
}

__global__ void k_pair_comb(const float* tn, const float* po, const float* ob,
                            float* tp, long n) {
  const long idx = (long)blockIdx.x * 256 + threadIdx.x;
  if (idx >= n) return;
  tp[idx] = tn[idx] + po[idx] + ob[idx % H2C];
}

__global__ void k_lambda(const float* lq1, const float* lk1,
                         const float* lq2, const float* lk2,
                         float lam_init, float* lam) {
  if (threadIdx.x == 0 && blockIdx.x == 0) {
    float d1 = 0.f, d2 = 0.f;
    for (int i = 0; i < HDD; ++i) { d1 += lq1[i]*lk1[i]; d2 += lq2[i]*lk2[i]; }
    lam[0] = expf(d1) - expf(d2) + lam_init;
  }
}

__global__ void k_colmean(const float* h, float* hm) {
  const int idx = blockIdx.x * 256 + threadIdx.x;
  if (idx >= BBATCH * DIMD) return;
  const int b = idx / DIMD, c = idx % DIMD;
  float s = 0.f;
  for (int i = 0; i < LL; ++i) s += h[((long)(b*LL + i)) * DIMD + c];
  hm[idx] = s * (1.f / (float)LL);
}

__global__ __launch_bounds__(128) void k_head(const float* hm, const float* g,
    const float* bta, const float* w, const float* hb, float* out) {
  const int b = threadIdx.x >> 5;
  const int lane = threadIdx.x & 31;
  const float* xr = hm + b * DIMD;
  float s1 = 0.f, s2 = 0.f;
  for (int i = lane; i < DIMD; i += 32) { const float v = xr[i]; s1 += v; s2 += v*v; }
  for (int o = 16; o > 0; o >>= 1) { s1 += __shfl_down(s1, o, 32); s2 += __shfl_down(s2, o, 32); }
  s1 = __shfl(s1, 0, 32); s2 = __shfl(s2, 0, 32);
  const float m = s1 / (float)DIMD;
  const float rs = rsqrtf(s2 / (float)DIMD - m*m + 1e-5f);
  float acc = 0.f;
  for (int i = lane; i < DIMD; i += 32) acc += ((xr[i]-m)*rs*g[i] + bta[i]) * w[i];
  for (int o = 16; o > 0; o >>= 1) acc += __shfl_down(acc, o, 32);
  if (lane == 0) out[b] = 1.f / (1.f + expf(-(acc + hb[0])));
}

// ----------------------------- host side -----------------------------
static inline void gemm_launch(hipStream_t st,
    const float* A, int lda, long aO, long aI,
    const float* Bm, int ldb, long bO, long bI, int transB,
    float* D, long dRow, int dCol, long dO, long dI,
    const float* bias, int M, int N, int K, int inner, int batch, float alpha) {
  GemmP p;
  p.A = A; p.Bm = Bm; p.D = D; p.bias = bias;
  p.aO = aO; p.aI = aI; p.bO = bO; p.bI = bI; p.dO = dO; p.dI = dI; p.dRow = dRow;
  p.inner = inner; p.lda = lda; p.ldb = ldb; p.dCol = dCol;
  p.M = M; p.N = N; p.K = K;
  p.hasBias = (bias != nullptr) ? 1 : 0; p.alpha = alpha;
  dim3 grid((M + 31) / 32, (N + 31) / 32, batch);
  const bool guard = (K % 32) != 0;
  if (transB) {
    if (guard) k_gemm<true,  true ><<<grid, dim3(32), 0, st>>>(p);
    else       k_gemm<true,  false><<<grid, dim3(32), 0, st>>>(p);
  } else {
    if (guard) k_gemm<false, true ><<<grid, dim3(32), 0, st>>>(p);
    else       k_gemm<false, false><<<grid, dim3(32), 0, st>>>(p);
  }
}

static inline dim3 ewg(long n) { return dim3((unsigned)((n + 255) / 256)); }
static inline dim3 rwg(long rows) { return dim3((unsigned)((rows + 7) / 8)); }

extern "C" void kernel_launch(void* const* d_in, const int* in_sizes, int n_in,
                              void* d_out, int out_size, void* d_ws, size_t ws_size,
                              hipStream_t stream) {
  (void)out_size; (void)ws_size;
  // ---- input index maps: insertion-order vs jax-pytree-sorted flattening
  const bool sorted = (n_in > 0) && (in_sizes[n_in - 1] == 5308416);
  int iX,iT,iC1W,iC1B,iC2W,iC2B,iFCW,iFCB,iEMW,iEMB,iPW1,iPB1,iPW2,iPB2,
      iPL,iPR,iHG,iHB,iHW,iHBS;
  int bWQ[6],bWK[6],bWV[6],bWO[6],bLQ1[6],bLK1[6],bLQ2[6],bLK2[6],
      bSG1[6],bSG2[6],bOG[6],bOBi[6],bOA[6],bOBW[6],bOW[6],bOOB[6];
  if (!sorted) {
    iX=0; iT=1; iC1W=2; iC1B=3; iC2W=4; iC2B=5; iFCW=6; iFCB=7; iEMW=8; iEMB=9;
    iPW1=10; iPB1=11; iPW2=12; iPB2=13; iPL=14; iPR=15; iHG=16; iHB=17; iHW=18; iHBS=19;
    for (int d = 0; d < 6; ++d) {
      const int b0 = 20 + 16*d;
      bWQ[d]=b0+0; bWK[d]=b0+1; bWV[d]=b0+2; bWO[d]=b0+3;
      bLQ1[d]=b0+4; bLK1[d]=b0+5; bLQ2[d]=b0+6; bLK2[d]=b0+7;
      bSG1[d]=b0+8; bSG2[d]=b0+9; bOG[d]=b0+10; bOBi[d]=b0+11;
      bOA[d]=b0+12; bOBW[d]=b0+13; bOW[d]=b0+14; bOOB[d]=b0+15;
    }
  } else {
    for (int d = 0; d < 6; ++d) {
      const int b0 = 16*d;
      bLK1[d]=b0+0; bLK2[d]=b0+1; bLQ1[d]=b0+2; bLQ2[d]=b0+3;
      bOA[d]=b0+4; bOBi[d]=b0+5; bOBW[d]=b0+6; bOG[d]=b0+7;
      bOOB[d]=b0+8; bOW[d]=b0+9; bSG1[d]=b0+10; bSG2[d]=b0+11;
      bWK[d]=b0+12; bWO[d]=b0+13; bWQ[d]=b0+14; bWV[d]=b0+15;
    }
    iC1B=96; iC1W=97; iC2B=98; iC2W=99; iEMB=100; iEMW=101; iFCB=102; iFCW=103;
    iHB=104; iHBS=105; iHG=106; iHW=107; iPL=108; iPR=109;
    iPB1=110; iPB2=111; iPW1=112; iPW2=113; iT=114; iX=115;
  }
  auto F = [&](int i) -> const float* { return (const float*)d_in[i]; };

  // ---- workspace bump allocator (all f32, 256B aligned)
  char* wsb = (char*)d_ws;
  size_t off = 0;
  auto alloc = [&](long nflt) -> float* {
    float* p = (float*)(wsb + off);
    off += (size_t)(((nflt * 4 + 255) / 256) * 256);
    return p;
  };
  float* bigreg = alloc(7077888);   // pooled (768x4096) then u (768x24x384)
  float* pooled = bigreg;
  float* u      = bigreg;
  float* hfc  = alloc(NTOKD);
  float* h    = alloc(NTOKD);
  float* tf   = alloc((long)BSR * FREQC);
  float* pos1 = alloc(NTOKD);
  float* pos2 = alloc(NTOKD);
  float* tp   = alloc(NPAIR);
  float* tn   = alloc(NPAIR);
  float* sm   = alloc(NPAIR);       // scores/softmax, then pair-gemm output
  float* attn = alloc(NATT);
  float* xb   = alloc(NTOKD);       // rms/ln scratch
  float* qb   = alloc(NTOKD);
  float* kb   = alloc(NTOKD);
  float* vb   = alloc(NTOKD);
  float* ob   = alloc(NTOKD);
  float* aob  = alloc(NTOKD);
  float* yb   = alloc(NTOKD);
  float* gb   = alloc(NTOKD);
  float* s1b  = alloc(NTOKD);
  float* ab   = alloc(NTOKD);
  float* bbv  = alloc(NTOKD);
  float* hm   = alloc(BBATCH * DIMD);
  float* lamw = alloc(8);

  // ---- conv stem (228KB dynamic LDS per workgroup)
  (void)hipFuncSetAttribute((const void*)k_conv,
      hipFuncAttributeMaxDynamicSharedMemorySize, CONV_SMEM_BYTES);
  k_conv<<<BSR, 256, CONV_SMEM_BYTES, stream>>>(F(iX), F(iC1W), F(iC1B),
                                                F(iC2W), F(iC2B), pooled);
  // fc + emb_seq
  gemm_launch(stream, pooled, 4096, 0,0, F(iFCW), 4096, 0,0, 1,
              hfc, DIMD,1, 0,0, F(iFCB), BSR, DIMD, 4096, 1, 1, 1.f);
  gemm_launch(stream, hfc, DIMD, 0,0, F(iEMW), DIMD, 0,0, 1,
              h, DIMD,1, 0,0, F(iEMB), BSR, DIMD, DIMD, 1, 1, 1.f);
  // timestep/pos embedding
  k_timestep<<<ewg((long)BSR*FREQC), 256, 0, stream>>>(F(iT), tf);
  gemm_launch(stream, tf, FREQC, 0,0, F(iPW1), FREQC, 0,0, 1,
              pos1, DIMD,1, 0,0, F(iPB1), BSR, DIMD, FREQC, 1, 1, 1.f);
  k_silu<<<ewg(NTOKD), 256, 0, stream>>>(pos1, NTOKD);
  gemm_launch(stream, pos1, DIMD, 0,0, F(iPW2), DIMD, 0,0, 1,
              pos2, DIMD,1, 0,0, F(iPB2), BSR, DIMD, DIMD, 1, 1, 1.f);
  k_add<<<ewg(NTOKD), 256, 0, stream>>>(h, h, pos2, NTOKD);
  // pair bias
  k_pair_init<<<ewg(NPAIR), 256, 0, stream>>>(F(iT), F(iPL), F(iPR), tp, NPAIR);

  for (int d = 0; d < 6; ++d) {
    const float lam_init = 0.8f - 0.6f * expf(-0.3f * (float)d);
    k_lambda<<<1, 32, 0, stream>>>(F(bLQ1[d]), F(bLK1[d]), F(bLQ2[d]), F(bLK2[d]),
                                   lam_init, lamw);
    // tn = rms(tp); xb = rms(h)
    k_rms<<<rwg((long)BBATCH*LL*LL), 256, 0, stream>>>(tp, tn, BBATCH*LL*LL, H2C, 1.f);
    k_rms<<<rwg(BSR), 256, 0, stream>>>(h, xb, BSR, DIMD, 1.f);
    // q,k,v projections
    gemm_launch(stream, xb, DIMD, 0,0, F(bWQ[d]), DIMD, 0,0, 1,
                qb, DIMD,1, 0,0, nullptr, BSR, DIMD, DIMD, 1, 1, 1.f);
    gemm_launch(stream, xb, DIMD, 0,0, F(bWK[d]), DIMD, 0,0, 1,
                kb, DIMD,1, 0,0, nullptr, BSR, DIMD, DIMD, 1, 1, 1.f);
    gemm_launch(stream, xb, DIMD, 0,0, F(bWV[d]), DIMD, 0,0, 1,
                vb, DIMD,1, 0,0, nullptr, BSR, DIMD, DIMD, 1, 1, 1.f);
    // scores: per (b, c2) 192x192, K=16 (zero-padded to 32), alpha = 1/sqrt(HD)
    gemm_launch(stream,
                qb, DIMD, (long)LL*DIMD, 16,
                kb, DIMD, (long)LL*DIMD, 16, 1,
                sm, LL, 1, (long)H2C*LL*LL, (long)LL*LL,
                nullptr, LL, LL, HDD, H2C, BBATCH*H2C, 0.25f);
    k_softmax<<<rwg((long)BBATCH*H2C*LL), 256, 0, stream>>>(sm, tn);
    k_diff<<<ewg(NATT), 256, 0, stream>>>(sm, lamw, attn, NATT);
    // o = attn @ v : per (b, head) 192x32, K=192
    gemm_launch(stream,
                attn, LL, (long)NHH*LL*LL, (long)LL*LL,
                vb, DIMD, (long)LL*DIMD, VDD, 0,
                ob, DIMD, 1, (long)LL*DIMD, VDD,
                nullptr, LL, VDD, LL, NHH, BBATCH*NHH, 1.f);
    // rms over head dim (32), scale (1 - lam_init), in place
    k_rms<<<rwg((long)BSR*NHH), 256, 0, stream>>>(ob, ob, BSR*NHH, VDD, 1.f - lam_init);
    gemm_launch(stream, ob, DIMD, 0,0, F(bWO[d]), DIMD, 0,0, 1,
                aob, DIMD,1, 0,0, nullptr, BSR, DIMD, DIMD, 1, 1, 1.f);
    k_add<<<ewg(NTOKD), 256, 0, stream>>>(yb, h, aob, NTOKD);
    // swiglu
    k_rms<<<rwg(BSR), 256, 0, stream>>>(yb, xb, BSR, DIMD, 1.f);
    gemm_launch(stream, xb, DIMD, 0,0, F(bSG2[d]), DIMD, 0,0, 1,
                gb, DIMD,1, 0,0, nullptr, BSR, DIMD, DIMD, 1, 1, 1.f);
    gemm_launch(stream, xb, DIMD, 0,0, F(bSG1[d]), DIMD, 0,0, 1,
                s1b, DIMD,1, 0,0, nullptr, BSR, DIMD, DIMD, 1, 1, 1.f);
    k_swiglu<<<ewg(NTOKD), 256, 0, stream>>>(yb, s1b, gb, h, NTOKD);
    // outer-product pair update
    k_ln<<<rwg(BSR), 256, 0, stream>>>(h, xb, F(bOG[d]), F(bOBi[d]), BSR, DIMD);
    gemm_launch(stream, xb, DIMD, 0,0, F(bOA[d]), DIMD, 0,0, 1,
                ab, DIMD,1, 0,0, nullptr, BSR, DIMD, DIMD, 1, 1, 1.f);
    gemm_launch(stream, xb, DIMD, 0,0, F(bOBW[d]), DIMD, 0,0, 1,
                bbv, DIMD,1, 0,0, nullptr, BSR, DIMD, DIMD, 1, 1, 1.f);
    // u[m, o, :] = a[m, :] @ W_o  (24 batched GEMMs, B non-transposed)
    gemm_launch(stream,
                ab, DIMD, 0, 0,
                F(bOW[d]), DIMD, 0, (long)DIMD*DIMD, 0,
                u, (long)H2C*DIMD, 1, 0, DIMD,
                nullptr, BSR, DIMD, DIMD, H2C, H2C, 1.f);
    // out[b,i,j,o] = u[b,i,o,:] @ bb[b,j,:]^T : per (b,o) 192x192, K=384
    gemm_launch(stream,
                u, H2C*DIMD, (long)LL*H2C*DIMD, DIMD,
                bbv, DIMD, (long)LL*DIMD, 0, 1,
                sm, (long)LL*H2C, H2C, (long)LL*LL*H2C, 1,
                nullptr, LL, LL, DIMD, H2C, BBATCH*H2C, 1.f);
    k_pair_comb<<<ewg(NPAIR), 256, 0, stream>>>(tn, sm, F(bOOB[d]), tp, NPAIR);
  }
  // head
  k_colmean<<<ewg((long)BBATCH*DIMD), 256, 0, stream>>>(h, hm);
  k_head<<<1, 128, 0, stream>>>(hm, F(iHG), F(iHB), F(iHW), F(iHBS), (float*)d_out);
}
